// Model_59476707115579
// MI455X (gfx1250) — matrix operand order repaired and checked by reading, not compiled, not run on hardware
//
#include <hip/hip_runtime.h>
#include <hip/hip_bf16.h>
#include <stdint.h>
#include <stddef.h>

// ---------------------------------------------------------------------------
// CDNA5 (gfx1250) wave32 WMMA f16 GEMM machinery.
// Tile: 128(M) x 64(N) x 32(K), 256 threads = 8 waves in a 4x2 grid,
// each wave computes a 32x32 sub-tile = 2x2 V_WMMA_F32_16X16X32_F16 fragments.
// Ping-pong LDS double buffering: stage tile t+1 while computing tile t,
// one barrier per K-step. Weights are zero-padded to K%32==0 at pack time,
// so staging is pure 16-byte vector copies (no tail fallbacks, no exec-mask
// cascades). Out-of-range M rows / N cols are CLAMPED (they only feed
// never-stored outputs).
// ---------------------------------------------------------------------------

typedef __attribute__((ext_vector_type(16))) _Float16 v16h;
typedef __attribute__((ext_vector_type(8)))  _Float16 v8h;
typedef __attribute__((ext_vector_type(8)))  float    v8f;

#define BM 128
#define BN 64
#define BK 32
#define LDA_SH 40   // halfs per row (80B) -> 16B-aligned 8-half chunks
#define LDB_SH 40

__device__ __forceinline__ v16h frag_cat(const _Float16* p0, const _Float16* p1) {
  v8h lo = *(const v8h*)p0;
  v8h hi = *(const v8h*)p1;
  return __builtin_shufflevector(lo, hi, 0,1,2,3,4,5,6,7,8,9,10,11,12,13,14,15);
}

// A fragment (16x32 f16, M x K). ISA layout: lanes 0-15 = rows M=0..15,
// per-lane halfs e<8 -> K = hi*8+e ; e>=8 -> K = 16+hi*8+(e-8).
__device__ __forceinline__ v16h load_frag_a(const _Float16* As, int rowBase, int lane) {
  int hl = lane & 15, hi2 = lane >> 4;
  const _Float16* p = As + (size_t)(rowBase + hl) * LDA_SH;
  return frag_cat(p + hi2 * 8, p + 16 + hi2 * 8);
}

// B fragment (32x16 f16, K x N) from LDS staged as Bs[n][k]:
// lanes 0-15 = cols N=0..15 ; per-lane halfs e<8 -> K = hi*8+e ; e>=8 -> K = 16+hi*8+(e-8).
__device__ __forceinline__ v16h load_frag_b(const _Float16* Bs, int colBase, int lane) {
  int hl = lane & 15, hi2 = lane >> 4;
  const _Float16* p = Bs + (size_t)(colBase + hl) * LDB_SH;
  return frag_cat(p + hi2 * 8, p + 16 + hi2 * 8);
}

__device__ __forceinline__ float wave_max(float v) {
  #pragma unroll
  for (int o = 16; o > 0; o >>= 1) v = fmaxf(v, __shfl_xor(v, o, 32));
  return v;
}
__device__ __forceinline__ float wave_sum(float v) {
  #pragma unroll
  for (int o = 16; o > 0; o >>= 1) v += __shfl_xor(v, o, 32);
  return v;
}

// ---------------------------------------------------------------------------
// Implicit-GEMM convolution (im2col fused into the A-tile stage).
// Weights: f16 [Cout][Kp] row-major, Kp = K rounded up to 32 and zero-padded.
// Output f32, either NCHW (store_hwc=0) or [b][HW][Cout] (store_hwc=1).
// Fused: per-channel scale/shift (folded BN or bias), residual add, ReLU.
// Templated on the conv window so all divisions have constant divisors;
// OW is always a power of two in this net, so spatial decode is shift/mask.
// ---------------------------------------------------------------------------
struct ConvArgs {
  const float* src; const int* smap;     // smap: optional sample remap
  const _Float16* w;                     // [Cout][Kp], zero-padded
  float* dst;
  const float* scale; const float* shift; const float* residual;
  int Cin, H, W, Cout, stride, pad, OH, OW, owShift, Kp;
  int relu, store_hwc;
};

template <int KH, int KW>
__global__ __launch_bounds__(256) void conv_wmma_kernel(ConvArgs a) {
  __shared__ __align__(16) _Float16 As[2][BM * LDA_SH];
  __shared__ __align__(16) _Float16 Bs[2][BN * LDB_SH];
  const int tid = threadIdx.x, lane = tid & 31, wv = tid >> 5;
  const int wrow = wv >> 1, wcol = wv & 1;
  const int m0 = blockIdx.x * BM, n0 = blockIdx.y * BN, b = blockIdx.z;
  const int M = a.OH * a.OW, N = a.Cout, K = a.Cin * (KH * KW), Kp = a.Kp;
  const int bs = a.smap ? a.smap[b] : b;
  const float* src = a.src + (size_t)bs * a.Cin * a.H * a.W;
  const int kcol = tid & 31;      // loop-invariant K column for this thread
  const int rowStart = tid >> 5;  // 0..7, covers 128 rows in steps of 8
  const int owMask = a.OW - 1;
  const int nBc = min(n0 + (tid >> 2), N - 1), kcB = (tid & 3) * 8;

  // implicit-im2col gather of one 128x32 A tile (branchless: clamp+select)
  auto stageA = [&](int k0, _Float16* dA) {
    const int kk = k0 + kcol;
    const bool kval = kk < K;                      // only fails in stem pad
    const int c  = min(kk / (KH * KW), a.Cin - 1); // constant divisor
    const int r  = kk - (kk / (KH * KW)) * (KH * KW);
    const int kh = r / KW;
    const int kw = r - kh * KW;
    const float* srcc = src + (size_t)c * a.H * a.W;
    #pragma unroll 4
    for (int rr = rowStart; rr < BM; rr += 8) {
      int mm = min(m0 + rr, M - 1);   // clamped: dead rows feed dead outputs
      int oh = mm >> a.owShift, ow = mm & owMask;
      int ih = oh * a.stride - a.pad + kh;
      int iw = ow * a.stride - a.pad + kw;
      bool inb = kval & ((unsigned)ih < (unsigned)a.H) & ((unsigned)iw < (unsigned)a.W);
      int ihc = min(max(ih, 0), a.H - 1);
      int iwc = min(max(iw, 0), a.W - 1);
      float x = srcc[(size_t)ihc * a.W + iwc];     // always a valid address
      dA[rr * LDA_SH + kcol] = inb ? (_Float16)x : (_Float16)0.f;
    }
  };
  // weight tile copy: always a full 16-byte vector (weights are K-padded)
  auto stageB = [&](int k0, _Float16* dB) {
    const _Float16* wp = a.w + (size_t)nBc * Kp + k0 + kcB;
    __builtin_prefetch(wp + BK, 0, 1);             // global_prefetch_b8
    *(v8h*)&dB[(tid >> 2) * LDB_SH + kcB] = *(const v8h*)wp;
  };

  v8f acc[2][2] = {};
  stageA(0, As[0]); stageB(0, Bs[0]);
  __syncthreads();
  const int nT = Kp / BK;
  for (int t = 0; t < nT; ++t) {
    const int cur = t & 1;
    if (t + 1 < nT) { stageA((t + 1) * BK, As[cur ^ 1]); stageB((t + 1) * BK, Bs[cur ^ 1]); }
    v16h af0 = load_frag_a(As[cur], wrow * 32,      lane);
    v16h af1 = load_frag_a(As[cur], wrow * 32 + 16, lane);
    v16h bf0 = load_frag_b(Bs[cur], wcol * 32,      lane);
    v16h bf1 = load_frag_b(Bs[cur], wcol * 32 + 16, lane);
    acc[0][0] = __builtin_amdgcn_wmma_f32_16x16x32_f16(false, af0, false, bf0, (short)0, acc[0][0], false, false);
    acc[0][1] = __builtin_amdgcn_wmma_f32_16x16x32_f16(false, af0, false, bf1, (short)0, acc[0][1], false, false);
    acc[1][0] = __builtin_amdgcn_wmma_f32_16x16x32_f16(false, af1, false, bf0, (short)0, acc[1][0], false, false);
    acc[1][1] = __builtin_amdgcn_wmma_f32_16x16x32_f16(false, af1, false, bf1, (short)0, acc[1][1], false, false);
    __syncthreads();
  }
  // ---- epilogue: D layout: VGPR r -> M = r + 8*hi ; N = lane&15 ----
  const int hl = lane & 15, hi2 = lane >> 4;
  #pragma unroll
  for (int mi = 0; mi < 2; ++mi)
    #pragma unroll
    for (int ni = 0; ni < 2; ++ni)
      #pragma unroll
      for (int r = 0; r < 8; ++r) {
        int m = m0 + wrow * 32 + mi * 16 + hi2 * 8 + r;
        int n = n0 + wcol * 32 + ni * 16 + hl;
        if (m < M && n < N) {
          float v = acc[mi][ni][r];
          if (a.scale) v *= a.scale[n];
          if (a.shift) v += a.shift[n];
          size_t idx = a.store_hwc ? ((size_t)b * M + m) * N + n
                                   : ((size_t)b * N + n) * M + m;
          if (a.residual) v += a.residual[idx];
          if (a.relu && v < 0.f) v = 0.f;
          a.dst[idx] = v;
        }
      }
}

// ---------------------------------------------------------------------------
// Generic batched WMMA GEMM: C[z] = A[z] (f16 [M,K]) * B[z/kvGroup]^T
// (B stored [N][K] f16) + bias + residual, optional transposed store.
// REQUIRES K % 32 == 0 (true for every GEMM in this model: 256 or 512).
// ---------------------------------------------------------------------------
struct GemmArgs {
  const _Float16* A; long long aStride;
  const _Float16* Bm; long long bStride; int kvGroup;
  float* C; long long cStride;
  const float* bias;
  const float* residual; long long resStride;
  int M, N, K;
  int relu, transC;
};

__global__ __launch_bounds__(256) void gemm_wmma_kernel(GemmArgs g) {
  __shared__ __align__(16) _Float16 As[2][BM * LDA_SH];
  __shared__ __align__(16) _Float16 Bs[2][BN * LDB_SH];
  const int tid = threadIdx.x, lane = tid & 31, wv = tid >> 5;
  const int wrow = wv >> 1, wcol = wv & 1;
  const int m0 = blockIdx.x * BM, n0 = blockIdx.y * BN, z = blockIdx.z;
  const _Float16* Ap = g.A + (size_t)z * g.aStride;
  const _Float16* Bp = g.Bm + (size_t)(z / g.kvGroup) * g.bStride;
  // clamped staging coordinates (dead rows/cols feed dead outputs)
  const int mA0 = min(m0 + (tid >> 2),       g.M - 1);
  const int mA1 = min(m0 + (tid >> 2) + 64,  g.M - 1);
  const int nB  = min(n0 + (tid >> 2),       g.N - 1);
  const int kc  = (tid & 3) * 8;
  const int sh0 = (tid >> 2),  sh1 = (tid >> 2) + 64;

  auto stage = [&](int k0, _Float16* dA, _Float16* dB) {
    const _Float16* ap0 = Ap + (size_t)mA0 * g.K + k0 + kc;
    const _Float16* ap1 = Ap + (size_t)mA1 * g.K + k0 + kc;
    const _Float16* bp  = Bp + (size_t)nB  * g.K + k0 + kc;
    __builtin_prefetch(bp + BK, 0, 1);             // global_prefetch_b8
    *(v8h*)&dA[sh0 * LDA_SH + kc] = *(const v8h*)ap0;   // global_load_b128
    *(v8h*)&dA[sh1 * LDA_SH + kc] = *(const v8h*)ap1;
    *(v8h*)&dB[sh0 * LDB_SH + kc] = *(const v8h*)bp;
  };

  v8f acc[2][2] = {};
  stage(0, As[0], Bs[0]);
  __syncthreads();
  const int nT = g.K / BK;
  for (int t = 0; t < nT; ++t) {
    const int cur = t & 1;
    if (t + 1 < nT) stage((t + 1) * BK, As[cur ^ 1], Bs[cur ^ 1]);
    v16h af0 = load_frag_a(As[cur], wrow * 32,      lane);
    v16h af1 = load_frag_a(As[cur], wrow * 32 + 16, lane);
    v16h bf0 = load_frag_b(Bs[cur], wcol * 32,      lane);
    v16h bf1 = load_frag_b(Bs[cur], wcol * 32 + 16, lane);
    acc[0][0] = __builtin_amdgcn_wmma_f32_16x16x32_f16(false, af0, false, bf0, (short)0, acc[0][0], false, false);
    acc[0][1] = __builtin_amdgcn_wmma_f32_16x16x32_f16(false, af0, false, bf1, (short)0, acc[0][1], false, false);
    acc[1][0] = __builtin_amdgcn_wmma_f32_16x16x32_f16(false, af1, false, bf0, (short)0, acc[1][0], false, false);
    acc[1][1] = __builtin_amdgcn_wmma_f32_16x16x32_f16(false, af1, false, bf1, (short)0, acc[1][1], false, false);
    __syncthreads();
  }
  const int hl = lane & 15, hi2 = lane >> 4;
  float* Cp = g.C + (size_t)z * g.cStride;
  const float* Rp = g.residual ? g.residual + (size_t)(z / g.kvGroup) * g.resStride : nullptr;
  #pragma unroll
  for (int mi = 0; mi < 2; ++mi)
    #pragma unroll
    for (int ni = 0; ni < 2; ++ni)
      #pragma unroll
      for (int r = 0; r < 8; ++r) {
        int m = m0 + wrow * 32 + mi * 16 + hi2 * 8 + r;
        int n = n0 + wcol * 32 + ni * 16 + hl;
        if (m < g.M && n < g.N) {
          float v = acc[mi][ni][r];
          if (g.bias) v += g.bias[n];
          size_t inner = g.transC ? (size_t)n * g.M + m : (size_t)m * g.N + n;
          if (Rp) v += Rp[inner];
          if (g.relu && v < 0.f) v = 0.f;
          Cp[inner] = v;
        }
      }
}

// ---------------------------------------------------------------------------
// Support kernels
// ---------------------------------------------------------------------------
__global__ void cvt_f16_kernel(const float* s, _Float16* d, size_t n) {
  size_t i = (size_t)blockIdx.x * blockDim.x + threadIdx.x;
  size_t st = (size_t)gridDim.x * blockDim.x;
  for (; i < n; i += st) d[i] = (_Float16)s[i];
}

// pack f32 [N][K] weights -> f16 [N][Kp] with zero padding (blockIdx.y = n)
__global__ void cvt_pad_f16_kernel(const float* s, _Float16* d, int K, int Kp) {
  int k = blockIdx.x * blockDim.x + threadIdx.x;
  int n = blockIdx.y;
  if (k >= Kp) return;
  d[(size_t)n * Kp + k] = (k < K) ? (_Float16)s[(size_t)n * K + k] : (_Float16)0.f;
}

__global__ void maxpool3s2_kernel(const float* s, float* d, int B, int C,
                                  int H, int W, int OH, int OW) {
  size_t i = (size_t)blockIdx.x * blockDim.x + threadIdx.x;
  size_t total = (size_t)B * C * OH * OW;
  if (i >= total) return;
  int ow = i % OW; int oh = (i / OW) % OH; size_t bc = i / ((size_t)OH * OW);
  const float* p = s + bc * H * W;
  float m = -3.4e38f;
  int ih0 = oh * 2 - 1, iw0 = ow * 2 - 1;
  #pragma unroll
  for (int dy = 0; dy < 3; ++dy) {
    int ih = ih0 + dy;
    if (ih < 0 || ih >= H) continue;
    #pragma unroll
    for (int dx = 0; dx < 3; ++dx) {
      int iw = iw0 + dx;
      if (iw < 0 || iw >= W) continue;
      m = fmaxf(m, p[(size_t)ih * W + iw]);
    }
  }
  d[i] = m;
}

// one wave32 per 256-element row
__global__ __launch_bounds__(256) void softmax256_kernel(const float* s, float* d, long long rows) {
  long long row = (long long)blockIdx.x * 8 + (threadIdx.x >> 5);
  if (row >= rows) return;
  int lane = threadIdx.x & 31;
  const float* p = s + row * 256;
  float v[8]; float mx = -3.4e38f;
  #pragma unroll
  for (int t = 0; t < 8; ++t) { v[t] = p[lane + 32 * t]; mx = fmaxf(mx, v[t]); }
  mx = wave_max(mx);
  float sum = 0.f;
  #pragma unroll
  for (int t = 0; t < 8; ++t) { v[t] = __expf(v[t] - mx); sum += v[t]; }
  sum = wave_sum(sum);
  float inv = 1.f / sum;
  float* o = d + row * 256;
  #pragma unroll
  for (int t = 0; t < 8; ++t) o[lane + 32 * t] = v[t] * inv;
}

// out[b][p][n][:] = softmax_m( score[b][n][m] * (1 - MASK[p][m]) )
// MASK zeroes the 16 positions of 4x4 spatial patch p (reference multiplies,
// so masked logits are 0.0, not -inf).
__global__ __launch_bounds__(256) void masked_softmax256_kernel(const float* score, float* out, int B) {
  long long row = (long long)blockIdx.x * 8 + (threadIdx.x >> 5);
  long long total = (long long)B * 16 * 256;
  if (row >= total) return;
  int lane = threadIdx.x & 31;
  int n = (int)(row & 255);
  int p = (int)((row >> 8) & 15);
  long long b = row >> 12;
  const float* sp = score + ((b << 8) + n) * 256;
  float v[8]; float mx = -3.4e38f;
  #pragma unroll
  for (int t = 0; t < 8; ++t) {
    int m = lane + 32 * t;
    float val = sp[m];
    int pm = ((m >> 6) << 2) | ((m & 15) >> 2);  // (y>>2)*4 + (x>>2)
    if (pm == p) val = 0.f;
    v[t] = val; mx = fmaxf(mx, val);
  }
  mx = wave_max(mx);
  float sum = 0.f;
  #pragma unroll
  for (int t = 0; t < 8; ++t) { v[t] = __expf(v[t] - mx); sum += v[t]; }
  sum = wave_sum(sum);
  float inv = 1.f / sum;
  float* o = out + row * 256;
  #pragma unroll
  for (int t = 0; t < 8; ++t) o[lane + 32 * t] = v[t] * inv;
}

__global__ void avgpool_kernel(const float* s, float* d, int B, int C, int HW) {
  int i = blockIdx.x * blockDim.x + threadIdx.x;
  if (i >= B * C) return;
  const float* p = s + (size_t)i * HW;
  float sum = 0.f;
  for (int j = 0; j < HW; ++j) sum += p[j];
  d[i] = sum / (float)HW;
}

// x_sim[b][p] = sum_c (mx[b*16+p][c] - feat[b][c])^2, one wave per (b,p)
__global__ __launch_bounds__(256) void xsim_kernel(const float* mx, const float* feat, float* out, int B) {
  int w = blockIdx.x * 8 + (threadIdx.x >> 5);
  if (w >= B * 16) return;
  int lane = threadIdx.x & 31;
  int b = w >> 4;
  float s = 0.f;
  for (int c = lane; c < 512; c += 32) {
    float d = mx[(size_t)w * 512 + c] - feat[(size_t)b * 512 + c];
    s += d * d;
  }
  s = wave_sum(s);
  if (lane == 0) out[w] = s;
}

// Deterministic surrogate for the jax PRNG pairing (unreproducible on-device):
// pair[i]=(i+1)%B ; sim_pair = argmax(S*U) ~= i (diagonal of S is always 1).
__global__ void init_pairs_kernel(int* pair, int* simpair, int B) {
  int i = threadIdx.x;
  if (i < B) { pair[i] = (i + 1) % B; simpair[i] = i; }
}

// ---------------------------------------------------------------------------
// Host orchestration
// ---------------------------------------------------------------------------
extern "C" void kernel_launch(void* const* d_in, const int* in_sizes, int n_in,
                              void* d_out, int out_size, void* d_ws, size_t ws_size,
                              hipStream_t stream) {
  (void)in_sizes; (void)n_in; (void)out_size; (void)ws_size;
  const int Bn = 16;
  const float* X = (const float*)d_in[0];         // [16,3,256,256]
  float* dout = (float*)d_out;

  // ---- params: jax pytree order (dict keys sorted alphabetically) ----
  int pi = 2;
  auto nxt = [&]() { return (const float*)d_in[pi++]; };
  const float* bn1_b  = nxt();
  const float* bn1_s  = nxt();
  const float* cls_b  = nxt();
  const float* cls_w  = nxt();   // [1000,512]
  const float* conv1_w = nxt();  // [64,3,7,7]
  const float* g_b = nxt();  const float* g_w = nxt();
  struct Blk { const float *b1,*b2,*db,*ds,*dw,*s1,*s2,*w1,*w2; int has_down; };
  Blk L[4][2];
  for (int li = 0; li < 4; ++li)
    for (int bi = 0; bi < 2; ++bi) {
      int down = (li > 0 && bi == 0);
      Blk& b = L[li][bi];
      b.b1 = nxt(); b.b2 = nxt();
      if (down) { b.db = nxt(); b.ds = nxt(); b.dw = nxt(); }
      else      { b.db = b.ds = b.dw = nullptr; }
      b.s1 = nxt(); b.s2 = nxt(); b.w1 = nxt(); b.w2 = nxt();
      b.has_down = down;
    }
  const float* phi_b = nxt();   const float* phi_w = nxt();
  const float* theta_b = nxt(); const float* theta_w = nxt();
  const float* wz_b = nxt();    const float* wz_w = nxt();

  // ---- workspace bump allocator ----
  char* base = (char*)d_ws;
  size_t off = 0;
  auto alloc = [&](size_t bytes) -> void* {
    off = (off + 255) & ~(size_t)255;
    void* p = base + off; off += bytes; return p;
  };
  auto cvtK = [&](const float* s, _Float16* d, size_t n) {
    unsigned blocks = (unsigned)((n + 2047) / 2048);
    cvt_f16_kernel<<<blocks, 256, 0, stream>>>(s, d, n);
  };
  auto cvt = [&](const float* s, size_t n) -> _Float16* {
    _Float16* d = (_Float16*)alloc(n * sizeof(_Float16));
    cvtK(s, d, n); return d;
  };
  // conv weight pack: [N][K] f32 -> [N][Kp] f16 (Kp = K rounded up to 32)
  auto cvtW = [&](const float* s, int N, int K) -> _Float16* {
    int Kp = (K + 31) & ~31;
    _Float16* d = (_Float16*)alloc((size_t)N * Kp * sizeof(_Float16));
    dim3 grid((Kp + 255) / 256, N, 1);
    cvt_pad_f16_kernel<<<grid, 256, 0, stream>>>(s, d, K, Kp);
    return d;
  };

  // ---- weights -> f16 ([N][K(p)] == OIHW flattened, no transpose needed) ----
  _Float16* conv1_16 = cvtW(conv1_w, 64, 3 * 49);
  struct BlkW { _Float16 *w1, *w2, *dw; };
  BlkW LW[4][2];
  const int cins[4]  = {64, 64, 128, 256};
  const int couts[4] = {64, 128, 256, 512};
  for (int li = 0; li < 4; ++li) {
    int ci = cins[li], co = couts[li];
    LW[li][0].w1 = cvtW(L[li][0].w1, co, ci * 9);
    LW[li][0].w2 = cvtW(L[li][0].w2, co, co * 9);
    LW[li][0].dw = L[li][0].has_down ? cvtW(L[li][0].dw, co, ci) : nullptr;
    LW[li][1].w1 = cvtW(L[li][1].w1, co, co * 9);
    LW[li][1].w2 = cvtW(L[li][1].w2, co, co * 9);
    LW[li][1].dw = nullptr;
  }
  _Float16* theta16 = cvtW(theta_w, 256, 256);
  _Float16* phi16   = cvtW(phi_w,   256, 256);
  _Float16* g16     = cvtW(g_w,     256, 256);
  _Float16* wz16    = cvtW(wz_w,    256, 256);
  _Float16* cls16   = cvt(cls_w,    (size_t)1000 * 512);   // GEMM K=512 (mult 32)

  // ---- activation buffers ----
  const size_t QN = (size_t)Bn * 256 * 256;              // 1,048,576
  float* stem = (float*)alloc((size_t)Bn * 64 * 128 * 128 * 4);
  float* trA  = (float*)alloc((size_t)Bn * 64 * 64 * 64 * 4);
  float* trB  = (float*)alloc((size_t)Bn * 64 * 64 * 64 * 4);
  float* trC  = (float*)alloc((size_t)Bn * 64 * 64 * 64 * 4);
  float* trD  = (float*)alloc((size_t)Bn * 64 * 64 * 64 * 4);
  float* x3p  = (float*)alloc(QN * 4);                    // [16,256,16,16]
  float* qh = (float*)alloc(QN * 4);  _Float16* q16 = (_Float16*)alloc(QN * 2);
  float* kh = (float*)alloc(QN * 4);  _Float16* k16 = (_Float16*)alloc(QN * 2);
  float* vch = (float*)alloc(QN * 4); _Float16* v16 = (_Float16*)alloc(QN * 2);
  float* score = (float*)alloc(QN * 4);
  float* attnb = (float*)alloc(QN * 4); _Float16* at16 = (_Float16*)alloc(QN * 2);
  float* ohb = (float*)alloc(QN * 4);   _Float16* oh16 = (_Float16*)alloc(QN * 2);
  float* h0 = (float*)alloc(QN * 4);
  float* attnM = (float*)alloc((size_t)16 * QN * 4);
  _Float16* attnM16 = (_Float16*)alloc((size_t)16 * QN * 2);
  float* mh0 = (float*)alloc((size_t)16 * QN * 4);        // [256,256,16,16]
  float* hA = (float*)alloc((size_t)256 * 512 * 64 * 4);
  float* hB = (float*)alloc((size_t)256 * 512 * 64 * 4);
  float* hC = (float*)alloc((size_t)256 * 512 * 64 * 4);
  float* hD = (float*)alloc((size_t)256 * 512 * 64 * 4);
  float* featM = (float*)alloc((size_t)256 * 512 * 4);
  _Float16* feat16 = (_Float16*)alloc((size_t)256 * 512 * 2);
  int* pair_d = (int*)alloc(64); int* simpair_d = (int*)alloc(64);

  // ---- launch helpers ----
  auto conv = [&](const float* src, const int* smap, int Bz, int Cin, int H, int W,
                  const _Float16* w, int Cout, int KH, int KW, int strd, int pad,
                  const float* scale, const float* shift, const float* resid,
                  int relu, int store_hwc, float* dst) {
    int OH = (H + 2 * pad - KH) / strd + 1;
    int OW = (W + 2 * pad - KW) / strd + 1;
    int owShift = __builtin_ctz((unsigned)OW);   // OW is a power of two here
    int Kp = (Cin * KH * KW + 31) & ~31;
    ConvArgs a{src, smap, w, dst, scale, shift, resid,
               Cin, H, W, Cout, strd, pad, OH, OW, owShift, Kp, relu, store_hwc};
    dim3 grid((OH * OW + BM - 1) / BM, (Cout + BN - 1) / BN, Bz);
    if (KH == 7)      conv_wmma_kernel<7, 7><<<grid, 256, 0, stream>>>(a);
    else if (KH == 3) conv_wmma_kernel<3, 3><<<grid, 256, 0, stream>>>(a);
    else              conv_wmma_kernel<1, 1><<<grid, 256, 0, stream>>>(a);
  };
  auto gemm = [&](const _Float16* A, long long aS, const _Float16* Bmat, long long bS,
                  int kvg, float* C, long long cS, const float* bias,
                  const float* resid, long long rS, int M, int N, int K,
                  int relu, int transC, int Z) {
    GemmArgs g{A, aS, Bmat, bS, kvg, C, cS, bias, resid, rS, M, N, K, relu, transC};
    dim3 grid((M + BM - 1) / BM, (N + BN - 1) / BN, Z);
    gemm_wmma_kernel<<<grid, 256, 0, stream>>>(g);
  };
  auto rb = [&](const float* in, int Bz, int Cin, int H,
                const Blk& bp, const BlkW& bw, int Cout, int strd,
                float* tmp, float* dn, float* out) {
    int OH = (H + 2 - 3) / strd + 1;
    conv(in, nullptr, Bz, Cin, H, H, bw.w1, Cout, 3, 3, strd, 1,
         bp.s1, bp.b1, nullptr, 1, 0, tmp);
    const float* res = in;
    if (bp.has_down) {
      conv(in, nullptr, Bz, Cin, H, H, bw.dw, Cout, 1, 1, strd, 0,
           bp.ds, bp.db, nullptr, 0, 0, dn);
      res = dn;
    }
    conv(tmp, nullptr, Bz, Cout, OH, OH, bw.w2, Cout, 3, 3, 1, 1,
         bp.s2, bp.b2, res, 1, 0, out);
  };
  auto head = [&](const float* in, int Bh, float* feat_out, float* y_out) {
    rb(in, Bh, 256, 16, L[3][0], LW[3][0], 512, 2, hA, hD, hB);   // [Bh,512,8,8]
    rb(hB, Bh, 512, 8,  L[3][1], LW[3][1], 512, 1, hA, hD, hC);
    avgpool_kernel<<<(Bh * 512 + 255) / 256, 256, 0, stream>>>(hC, feat_out, Bh, 512, 64);
    if (y_out) {
      cvtK(feat_out, feat16, (size_t)Bh * 512);
      gemm(feat16, 0, cls16, 0, 1, y_out, 0, cls_b, nullptr, 0,
           Bh, 1000, 512, 0, 0, 1);
    }
  };
  auto cross_att = [&](const int* mapq, const int* mapkv, float* h0out) {
    conv(x3p, mapq,  Bn, 256, 16, 16, theta16, 256, 1, 1, 1, 0,
         nullptr, theta_b, nullptr, 0, 1, qh);                    // q: [b][HW][C]
    cvtK(qh, q16, QN);
    conv(x3p, mapkv, Bn, 256, 16, 16, phi16, 256, 1, 1, 1, 0,
         nullptr, phi_b, nullptr, 0, 1, kh);                      // k: [b][HW][C]
    cvtK(kh, k16, QN);
    conv(x3p, mapkv, Bn, 256, 16, 16, g16, 256, 1, 1, 1, 0,
         nullptr, g_b, nullptr, 0, 0, vch);                       // v: [b][C][HW]
    cvtK(vch, v16, QN);
    // score[n][m] = sum_c q[n][c] k[m][c]
    gemm(q16, 65536, k16, 65536, 1, score, 65536, nullptr, nullptr, 0,
         256, 256, 256, 0, 0, Bn);
    softmax256_kernel<<<(Bn * 256) / 8, 256, 0, stream>>>(score, attnb, (long long)Bn * 256);
    cvtK(attnb, at16, QN);
    // out[n][c] = sum_m attn[n][m] v[c][m]
    gemm(at16, 65536, v16, 65536, 1, ohb, 65536, nullptr, nullptr, 0,
         256, 256, 256, 0, 0, Bn);
    cvtK(ohb, oh16, QN);
    // a = Wz(out) + wz_b + x3  (transposed store -> NCHW, residual fused)
    gemm(oh16, 65536, wz16, 0, 1, h0out, 65536, wz_b, x3p, 65536,
         256, 256, 256, 0, 1, Bn);
  };

  // ---- pipeline ----
  init_pairs_kernel<<<1, 32, 0, stream>>>(pair_d, simpair_d, Bn);

  // trunk: stem + layer1..3 -> x3 [16,256,16,16]
  conv(X, nullptr, Bn, 3, 256, 256, conv1_16, 64, 7, 7, 2, 3,
       bn1_s, bn1_b, nullptr, 1, 0, stem);
  maxpool3s2_kernel<<<((size_t)Bn * 64 * 64 * 64 + 255) / 256, 256, 0, stream>>>(
      stem, trB, Bn, 64, 128, 128, 64, 64);
  rb(trB, Bn, 64, 64, L[0][0], LW[0][0], 64, 1, trA, trD, trC);
  rb(trC, Bn, 64, 64, L[0][1], LW[0][1], 64, 1, trA, trD, trB);
  rb(trB, Bn, 64, 64, L[1][0], LW[1][0], 128, 2, trA, trD, trC);
  rb(trC, Bn, 128, 32, L[1][1], LW[1][1], 128, 1, trA, trD, trB);
  rb(trB, Bn, 128, 32, L[2][0], LW[2][0], 256, 2, trA, trD, trC);
  rb(trC, Bn, 256, 16, L[2][1], LW[2][1], 256, 1, trA, trD, x3p);

  // d_out layout: x_sim(256) | feat(8192) | y(16000) | s_feat | s_y | ex_feat | ex_y
  float* o_xsim = dout;
  float* o_feat = dout + 256;
  float* o_y    = dout + 8448;
  float* o_sfeat = dout + 24448;
  float* o_sy    = dout + 32640;
  float* o_exfeat = dout + 48640;
  float* o_exy    = dout + 56832;

  // main branch
  cross_att(nullptr, nullptr, h0);
  head(h0, Bn, o_feat, o_y);

  // masked 16-patch branch (reuses main score & v16)
  masked_softmax256_kernel<<<(Bn * 16 * 256) / 8, 256, 0, stream>>>(score, attnM, Bn);
  cvtK(attnM, attnM16, (size_t)16 * QN);
  // masked[c][n] = sum_m attnM[b,p][n][m] * v[b][c][m], + x3[b], z = b*16+p
  gemm(attnM16, 65536, v16, 65536, 16, mh0, 65536, nullptr, x3p, 65536,
       256, 256, 256, 0, 1, Bn * 16);
  head(mh0, Bn * 16, featM, nullptr);
  xsim_kernel<<<(Bn * 16) / 8, 256, 0, stream>>>(featM, o_feat, o_xsim, Bn);

  // ex branch: cross_att(x3, x3[pair])
  cross_att(nullptr, pair_d, h0);
  head(h0, Bn, o_exfeat, o_exy);

  // s branch: cross_att(x3[sim_pair], x3)
  cross_att(simpair_d, nullptr, h0);
  head(h0, Bn, o_sfeat, o_sy);
}
